// InternImageLayer_34316788695739
// MI455X (gfx1250) — matrix operand order repaired and checked by reading, not compiled, not run on hardware
//
#include <hip/hip_runtime.h>
#include <hip/hip_bf16.h>
#include <math.h>

typedef __attribute__((ext_vector_type(16))) _Float16 v16h;
typedef __attribute__((ext_vector_type(8)))  _Float16 v8h;
typedef __attribute__((ext_vector_type(4)))  _Float16 v4h;
typedef __attribute__((ext_vector_type(8)))  float    v8f;

#define L_TOT 65536   // N*H*W
#define C_CH  128
#define H_DIM 128
#define W_DIM 128
#define G_GRP 4
#define HW_SZ 16384   // H*W
#define OM_LD 128     // padded row stride of offset+mask logits

// ---------------- wave reduction (wave32) ----------------
__device__ __forceinline__ float wave_sum32(float v) {
    v += __shfl_xor(v, 16, 32);
    v += __shfl_xor(v, 8, 32);
    v += __shfl_xor(v, 4, 32);
    v += __shfl_xor(v, 2, 32);
    v += __shfl_xor(v, 1, 32);
    return v;
}

__device__ __forceinline__ float gelu_exact(float v) {
    return 0.5f * v * (1.0f + erff(v * 0.70710678118654752f));
}

// ---------------- weight prep ----------------
__global__ void cvt_f32_to_f16(const float* __restrict__ src, _Float16* __restrict__ dst, int n) {
    int i = blockIdx.x * blockDim.x + threadIdx.x;
    if (i < n) dst[i] = (_Float16)src[i];
}

// pack off_w(72x128) + mask_w(36x128) + 20 zero rows -> 128x128 f16 ; biases -> 128 f32
__global__ void pack_offmask(const float* __restrict__ off_w, const float* __restrict__ mask_w,
                             const float* __restrict__ off_b, const float* __restrict__ mask_b,
                             _Float16* __restrict__ wp, float* __restrict__ bp) {
    int i = blockIdx.x * blockDim.x + threadIdx.x;
    if (i < OM_LD * 128) {
        int r = i >> 7, c = i & 127;
        float v = (r < 72) ? off_w[r * 128 + c] : ((r < 108) ? mask_w[(r - 72) * 128 + c] : 0.0f);
        wp[i] = (_Float16)v;
    }
    if (i < OM_LD) bp[i] = (i < 72) ? off_b[i] : ((i < 108) ? mask_b[i - 72] : 0.0f);
}

// ---------------- LN1: x (NCHW f32) -> xcl (NHWC f32) ----------------
__global__ void ln1_kernel(const float* __restrict__ x, const float* __restrict__ w,
                           const float* __restrict__ b, float* __restrict__ xcl) {
    int wid = (blockIdx.x * blockDim.x + threadIdx.x) >> 5;   // token index
    int lane = threadIdx.x & 31;
    int n = wid >> 14, hw = wid & (HW_SZ - 1);
    const float* xp = x + (size_t)n * C_CH * HW_SZ + hw;
    int c0 = lane * 4;
    float v[4];
#pragma unroll
    for (int j = 0; j < 4; ++j) v[j] = xp[(size_t)(c0 + j) * HW_SZ];
    float s = v[0] + v[1] + v[2] + v[3];
    float sq = v[0]*v[0] + v[1]*v[1] + v[2]*v[2] + v[3]*v[3];
    s = wave_sum32(s); sq = wave_sum32(sq);
    float m = s * (1.0f / 128.0f);
    float var = sq * (1.0f / 128.0f) - m * m;
    float rs = rsqrtf(var + 1e-6f);
    float4 o;
    o.x = (v[0] - m) * rs * w[c0 + 0] + b[c0 + 0];
    o.y = (v[1] - m) * rs * w[c0 + 1] + b[c0 + 1];
    o.z = (v[2] - m) * rs * w[c0 + 2] + b[c0 + 2];
    o.w = (v[3] - m) * rs * w[c0 + 3] + b[c0 + 3];
    *(float4*)(xcl + (size_t)wid * C_CH + c0) = o;
}

// ------- depthwise 3x3 conv (pad=1) + LN + GELU : xcl -> x1 (f32 + f16) -------
__global__ void dw_ln_gelu_kernel(const float* __restrict__ xcl, const float* __restrict__ dww,
                                  const float* __restrict__ dwb, const float* __restrict__ lw,
                                  const float* __restrict__ lb, float* __restrict__ x1f,
                                  _Float16* __restrict__ x1h) {
    int wid = (blockIdx.x * blockDim.x + threadIdx.x) >> 5;
    int lane = threadIdx.x & 31;
    int n = wid >> 14, h = (wid >> 7) & 127, w = wid & 127;
    int c0 = lane * 4;
    float acc[4] = {dwb[c0], dwb[c0 + 1], dwb[c0 + 2], dwb[c0 + 3]};
#pragma unroll
    for (int i = 0; i < 3; ++i) {
        int hh = h + i - 1;
        if (hh < 0 || hh >= H_DIM) continue;
#pragma unroll
        for (int j = 0; j < 3; ++j) {
            int ww = w + j - 1;
            if (ww < 0 || ww >= W_DIM) continue;
            float4 xv = *(const float4*)(xcl + ((size_t)n * HW_SZ + hh * W_DIM + ww) * C_CH + c0);
            const float* xvp = (const float*)&xv;
#pragma unroll
            for (int ch = 0; ch < 4; ++ch)
                acc[ch] += dww[(c0 + ch) * 9 + i * 3 + j] * xvp[ch];
        }
    }
    float s = acc[0] + acc[1] + acc[2] + acc[3];
    float sq = acc[0]*acc[0] + acc[1]*acc[1] + acc[2]*acc[2] + acc[3]*acc[3];
    s = wave_sum32(s); sq = wave_sum32(sq);
    float m = s * (1.0f / 128.0f);
    float var = sq * (1.0f / 128.0f) - m * m;
    float rs = rsqrtf(var + 1e-6f);
    float4 o; float* op = (float*)&o; v4h oh;
#pragma unroll
    for (int ch = 0; ch < 4; ++ch) {
        float g = gelu_exact((acc[ch] - m) * rs * lw[c0 + ch] + lb[c0 + ch]);
        op[ch] = g; oh[ch] = (_Float16)g;
    }
    *(float4*)(x1f + (size_t)wid * C_CH + c0) = o;
    *(v4h*)(x1h + (size_t)wid * C_CH + c0) = oh;
}

// ---------------- WMMA fragment loads ----------------
__device__ __forceinline__ v16h frag_a16(const _Float16* A, int ldk, int row, int k0, int lane) {
    int hi = lane >> 4;
    const _Float16* p = A + (size_t)row * ldk + k0 + hi * 8;
    v8h lo = *(const v8h*)p;
    v8h hh = *(const v8h*)(p + 16);
    return __builtin_shufflevector(lo, hh, 0,1,2,3,4,5,6,7,8,9,10,11,12,13,14,15);
}
__device__ __forceinline__ v16h frag_b16(const _Float16* W, int ldk, int col, int k0, int lane) {
    int hi = lane >> 4;
    const _Float16* p = W + (size_t)col * ldk + k0 + hi * 8;
    v8h lo = *(const v8h*)p;
    v8h hh = *(const v8h*)(p + 16);
    return __builtin_shufflevector(lo, hh, 0,1,2,3,4,5,6,7,8,9,10,11,12,13,14,15);
}

// ---- WMMA GEMM: C(MxN) = A(MxK) @ W(NxK)^T + bias; wave owns a 16x64 strip ----
// N must be a multiple of 64. Epilogue by template:
// MODE 0: store f32 row-major (off/mask logits)
// MODE 1: += x(NCHW) residual, store f32 row-major (xres)
// MODE 2: GELU, store f16 row-major (fc1 hidden)
// MODE 3: += xres(row-major), store f32 NCHW (final output)
template <int MODE>
__global__ void gemm_wmma_f16(const _Float16* __restrict__ A, const _Float16* __restrict__ W,
                              const float* __restrict__ bias, const float* __restrict__ xin,
                              float* __restrict__ outf, _Float16* __restrict__ outh,
                              int K, int Nn) {
    int wid = (blockIdx.x * blockDim.x + threadIdx.x) >> 5;
    int lane = threadIdx.x & 31;
    const int mtiles = L_TOT >> 4;
    int mt = wid % mtiles;
    int ng = wid / mtiles;
    int hi = lane >> 4;
    int ln = lane & 15;
    int row = mt * 16 + ln;
    int nbase = ng * 64;
    v8f acc0 = {}, acc1 = {}, acc2 = {}, acc3 = {};
#pragma unroll 2
    for (int k0 = 0; k0 < K; k0 += 32) {
        v16h a = frag_a16(A, K, row, k0, lane);
        v16h b0 = frag_b16(W, K, nbase + 0 * 16 + ln, k0, lane);
        v16h b1 = frag_b16(W, K, nbase + 1 * 16 + ln, k0, lane);
        v16h b2 = frag_b16(W, K, nbase + 2 * 16 + ln, k0, lane);
        v16h b3 = frag_b16(W, K, nbase + 3 * 16 + ln, k0, lane);
        acc0 = __builtin_amdgcn_wmma_f32_16x16x32_f16(false, a, false, b0, (short)0, acc0, false, false);
        acc1 = __builtin_amdgcn_wmma_f32_16x16x32_f16(false, a, false, b1, (short)0, acc1, false, false);
        acc2 = __builtin_amdgcn_wmma_f32_16x16x32_f16(false, a, false, b2, (short)0, acc2, false, false);
        acc3 = __builtin_amdgcn_wmma_f32_16x16x32_f16(false, a, false, b3, (short)0, acc3, false, false);
    }
    v8f accs[4] = {acc0, acc1, acc2, acc3};
#pragma unroll
    for (int s = 0; s < 4; ++s) {
        int col = nbase + s * 16 + ln;
        float bv = bias[col];
#pragma unroll
        for (int r = 0; r < 8; ++r) {
            int rr = mt * 16 + r + hi * 8;
            float v = accs[s][r] + bv;
            if constexpr (MODE == 0) {
                outf[(size_t)rr * OM_LD + col] = v;
            } else if constexpr (MODE == 1) {
                int n = rr >> 14, hw = rr & (HW_SZ - 1);
                v += xin[(((size_t)n * C_CH + col) << 14) + hw];
                outf[(size_t)rr * C_CH + col] = v;
            } else if constexpr (MODE == 2) {
                outh[(size_t)rr * Nn + col] = (_Float16)gelu_exact(v);
            } else {
                v += xin[(size_t)rr * C_CH + col];
                int n = rr >> 14, hw = rr & (HW_SZ - 1);
                outf[(((size_t)n * C_CH + col) << 14) + hw] = v;
            }
        }
    }
}

// ---------------- softmax over 9 taps per (token, group), in place ----------------
__global__ void softmax_mask_kernel(float* __restrict__ om) {
    int i = blockIdx.x * blockDim.x + threadIdx.x;   // token*4 + g
    if (i >= L_TOT * G_GRP) return;
    int g = i & 3;
    float* p = om + (size_t)(i >> 2) * OM_LD + 72 + g * 9;
    float mx = -1e30f;
#pragma unroll
    for (int k = 0; k < 9; ++k) mx = fmaxf(mx, p[k]);
    float s = 0.0f; float e[9];
#pragma unroll
    for (int k = 0; k < 9; ++k) { e[k] = expf(p[k] - mx); s += e[k]; }
    float inv = 1.0f / s;
#pragma unroll
    for (int k = 0; k < 9; ++k) p[k] = e[k] * inv;
}

// -------- DCN core: bilinear deformable sampling; wave = (token, group), lane = channel --------
__global__ void dcn_core_kernel(const float* __restrict__ x1, const float* __restrict__ om,
                                _Float16* __restrict__ dout) {
    int wid = (blockIdx.x * blockDim.x + threadIdx.x) >> 5;
    int lane = threadIdx.x & 31;
    int g = wid & 3;
    int pos = wid >> 2;
    int n = pos >> 14, h = (pos >> 7) & 127, w = pos & 127;
    const float* omr = om + (size_t)pos * OM_LD;
    int ch = g * 32 + lane;
    const float* base = x1 + (size_t)n * HW_SZ * C_CH + ch;
    float acc = 0.0f;
#pragma unroll
    for (int p = 0; p < 9; ++p) {
        float offx = omr[(g * 9 + p) * 2 + 0];
        float offy = omr[(g * 9 + p) * 2 + 1];
        float mval = omr[72 + g * 9 + p];
        int dx = p / 3 - 1, dy = p % 3 - 1;
        float gx = (float)(w + dx) + offx;          // unpadded coords (padded border is zero)
        float gy = (float)(h + dy) + offy;
        float x0f = floorf(gx), y0f = floorf(gy);
        float wx = gx - x0f, wy = gy - y0f;
        int x0 = (int)x0f, y0 = (int)y0f;
        float v00 = 0.f, v01 = 0.f, v10 = 0.f, v11 = 0.f;
        bool xa = (x0 >= 0) & (x0 < W_DIM), xb = (x0 + 1 >= 0) & (x0 + 1 < W_DIM);
        bool ya = (y0 >= 0) & (y0 < H_DIM), yb = (y0 + 1 >= 0) & (y0 + 1 < H_DIM);
        if (ya & xa) v00 = base[((size_t)y0 * W_DIM + x0) * C_CH];
        if (ya & xb) v01 = base[((size_t)y0 * W_DIM + x0 + 1) * C_CH];
        if (yb & xa) v10 = base[((size_t)(y0 + 1) * W_DIM + x0) * C_CH];
        if (yb & xb) v11 = base[((size_t)(y0 + 1) * W_DIM + x0 + 1) * C_CH];
        float bil = v00 * (1.f - wx) * (1.f - wy) + v01 * wx * (1.f - wy)
                  + v10 * (1.f - wx) * wy + v11 * wx * wy;
        acc += mval * bil;
    }
    dout[(size_t)pos * C_CH + ch] = (_Float16)acc;
}

// ---------------- LN2: xres (row-major f32) -> yh (row-major f16) ----------------
__global__ void ln2_kernel(const float* __restrict__ xres, const float* __restrict__ w,
                           const float* __restrict__ b, _Float16* __restrict__ yh) {
    int wid = (blockIdx.x * blockDim.x + threadIdx.x) >> 5;
    int lane = threadIdx.x & 31;
    int c0 = lane * 4;
    float4 xv = *(const float4*)(xres + (size_t)wid * C_CH + c0);
    const float* v = (const float*)&xv;
    float s = v[0] + v[1] + v[2] + v[3];
    float sq = v[0]*v[0] + v[1]*v[1] + v[2]*v[2] + v[3]*v[3];
    s = wave_sum32(s); sq = wave_sum32(sq);
    float m = s * (1.0f / 128.0f);
    float var = sq * (1.0f / 128.0f) - m * m;
    float rs = rsqrtf(var + 1e-6f);
    v4h o;
#pragma unroll
    for (int ch = 0; ch < 4; ++ch)
        o[ch] = (_Float16)((v[ch] - m) * rs * w[c0 + ch] + b[c0 + ch]);
    *(v4h*)(yh + (size_t)wid * C_CH + c0) = o;
}

extern "C" void kernel_launch(void* const* d_in, const int* in_sizes, int n_in,
                              void* d_out, int out_size, void* d_ws, size_t ws_size,
                              hipStream_t stream) {
    const float* x        = (const float*)d_in[0];
    const float* norm1_w  = (const float*)d_in[1];
    const float* norm1_b  = (const float*)d_in[2];
    const float* dw_w     = (const float*)d_in[3];
    const float* dw_b     = (const float*)d_in[4];
    const float* dwln_w   = (const float*)d_in[5];
    const float* dwln_b   = (const float*)d_in[6];
    const float* off_w    = (const float*)d_in[7];
    const float* off_b    = (const float*)d_in[8];
    const float* mask_w   = (const float*)d_in[9];
    const float* mask_b   = (const float*)d_in[10];
    const float* proj_w   = (const float*)d_in[11];
    const float* proj_b   = (const float*)d_in[12];
    const float* norm2_w  = (const float*)d_in[13];
    const float* norm2_b  = (const float*)d_in[14];
    const float* fc1_w    = (const float*)d_in[15];
    const float* fc1_b    = (const float*)d_in[16];
    const float* fc2_w    = (const float*)d_in[17];
    const float* fc2_b    = (const float*)d_in[18];

    // workspace layout with lifetime reuse
    char* p = (char*)d_ws;
    float*    xcl   = (float*)p;                     // steps 1-2 (32 MB of this 64 MB region)
    _Float16* hbuf  = (_Float16*)p;                  // fc1->fc2 hidden (64 MB: L x 512 f16)
    p += (size_t)64 << 20;
    float*    x1f   = (float*)p;                     // steps 2-5
    float*    xres  = (float*)p;                     // steps 6-9 (same region)
    p += (size_t)L_TOT * C_CH * 4;
    _Float16* x1h   = (_Float16*)p;                  // off/mask GEMM input
    _Float16* dcnh  = (_Float16*)p;                  // dcn out (same region, later)
    _Float16* yh    = (_Float16*)p;                  // ln2 out (same region, later)
    p += (size_t)L_TOT * C_CH * 2;
    float*    om    = (float*)p;                     // L x 128 (padded) logits / masks
    p += (size_t)L_TOT * OM_LD * 4;
    _Float16* wpack = (_Float16*)p;  p += OM_LD * 128 * 2;
    float*    bpack = (float*)p;     p += 1024;
    _Float16* wproj = (_Float16*)p;  p += 128 * 128 * 2;
    _Float16* wfc1  = (_Float16*)p;  p += 512 * 128 * 2;
    _Float16* wfc2  = (_Float16*)p;  p += 128 * 512 * 2;

    // --- weight prep (deterministic, every call) ---
    pack_offmask<<<(OM_LD * 128 + 255) / 256, 256, 0, stream>>>(off_w, mask_w, off_b, mask_b, wpack, bpack);
    cvt_f32_to_f16<<<(128 * 128 + 255) / 256, 256, 0, stream>>>(proj_w, wproj, 128 * 128);
    cvt_f32_to_f16<<<(512 * 128 + 255) / 256, 256, 0, stream>>>(fc1_w, wfc1, 512 * 128);
    cvt_f32_to_f16<<<(128 * 512 + 255) / 256, 256, 0, stream>>>(fc2_w, wfc2, 128 * 512);

    const int tokBlocks = L_TOT * 32 / 256;          // wave-per-token kernels

    // 1) LN1
    ln1_kernel<<<tokBlocks, 256, 0, stream>>>(x, norm1_w, norm1_b, xcl);
    // 2) depthwise conv + LN + GELU
    dw_ln_gelu_kernel<<<tokBlocks, 256, 0, stream>>>(xcl, dw_w, dw_b, dwln_w, dwln_b, x1f, x1h);

    const int mtiles = L_TOT / 16;
    // 3) offset+mask GEMM (N padded to 128)
    gemm_wmma_f16<0><<<mtiles * 2 / 8, 256, 0, stream>>>(x1h, wpack, bpack, nullptr, om, nullptr, 128, 128);
    // 4) softmax masks
    softmax_mask_kernel<<<L_TOT * G_GRP / 256, 256, 0, stream>>>(om);
    // 5) deformable sampling -> dcn out (f16)
    dcn_core_kernel<<<L_TOT * G_GRP * 32 / 256, 256, 0, stream>>>(x1f, om, dcnh);
    // 6) out-projection + residual with original x -> xres (f32 row-major)
    gemm_wmma_f16<1><<<mtiles * 2 / 8, 256, 0, stream>>>(dcnh, wproj, proj_b, x, xres, nullptr, 128, 128);
    // 7) LN2 -> f16
    ln2_kernel<<<tokBlocks, 256, 0, stream>>>(xres, norm2_w, norm2_b, yh);
    // 8) fc1 + GELU (N=512)
    gemm_wmma_f16<2><<<mtiles * 8 / 8, 256, 0, stream>>>(yh, wfc1, fc1_b, nullptr, nullptr, hbuf, 128, 512);
    // 9) fc2 + residual, store NCHW fp32 output
    gemm_wmma_f16<3><<<mtiles * 2 / 8, 256, 0, stream>>>(hbuf, wfc2, fc2_b, xres, (float*)d_out, nullptr, 512, 128);
}
